// GGNNClassifierFeatsNoEmb_66254165508835
// MI455X (gfx1250) — matrix operand order repaired and checked by reading, not compiled, not run on hardware
//
#include <hip/hip_runtime.h>
#include <hip/hip_bf16.h>

#define NUM_TYPES 60
#define DIM_TOK   129
#define SMALL_F   2
#define C         191
#define CP        192
#define C3P       576
#define STEPS     10
#define BLOCKS    5
#define NT        3
#define NN        4000
#define EE        12000
#define GG        32
#define LN_EPS    1e-5f

#define ROWTILES  (NN / 16)          // 250
#define ROWGRPS   ((ROWTILES + 7)/8) // 32 blocks of 8 waves
#define BSTRIDE   200                // LDS column stride in halves (400B, 16B aligned)

typedef __attribute__((ext_vector_type(16))) __bf16 v16bf;
typedef __attribute__((ext_vector_type(8)))  float  v8f;

// ---------------------------------------------------------------- utilities
__global__ void k_fill(float* __restrict__ p, int n, float v) {
  int i = blockIdx.x * blockDim.x + threadIdx.x;
  if (i < n) p[i] = v;
}
__global__ void k_copy(float* __restrict__ d, const float* __restrict__ s, int n) {
  int i = blockIdx.x * blockDim.x + threadIdx.x;
  if (i < n) d[i] = s[i];
}
__global__ void k_add(float* __restrict__ d, const float* __restrict__ s, int n) {
  int i = blockIdx.x * blockDim.x + threadIdx.x;
  if (i < n) d[i] += s[i];
}

// ---------------------------------------------------------------- features
__global__ void k_build(const int* __restrict__ x_type, const int* __restrict__ x_tok,
                        const float* __restrict__ x_small, float* __restrict__ h) {
  int n = blockIdx.x, c = threadIdx.x;          // blockDim = 192
  float v = 0.f;
  if (c < NUM_TYPES) {
    v = (x_type[n] == c) ? 1.f : 0.f;
  } else if (c < NUM_TYPES + DIM_TOK) {
    int tk = x_tok[n]; tk = tk < 0 ? 0 : (tk > DIM_TOK - 1 ? DIM_TOK - 1 : tk);
    v = (tk == c - NUM_TYPES) ? 1.f : 0.f;
  } else if (c < C) {
    v = x_small[n * SMALL_F + (c - NUM_TYPES - DIM_TOK)];
  }
  h[n * CP + c] = v;
}

// ---------------------------------------------------------------- weight converts
// conv_w [150][191][191] fp32 row-major  ->  [150][192 cols][192 k] bf16 column-major
__global__ void k_cvt_convw(const float* __restrict__ w, __bf16* __restrict__ out) {
  int tid = blockIdx.x * blockDim.x + threadIdx.x;
  if (tid >= 150 * CP * CP) return;
  int k = tid % CP, j = (tid / CP) % CP, s = tid / (CP * CP);
  float v = (j < C && k < C) ? w[(size_t)s * C * C + k * C + j] : 0.f;  // B[k][j]=W[k][j]
  out[tid] = (__bf16)v;
}
// gru_w [15][573][191] fp32 -> [15][3][192 rows j][192 k] bf16 (row j = wih row g*191+j)
__global__ void k_cvt_gruw(const float* __restrict__ w, __bf16* __restrict__ out) {
  int tid = blockIdx.x * blockDim.x + threadIdx.x;
  if (tid >= 15 * 3 * CP * CP) return;
  int k = tid % CP, j = (tid / CP) % CP, g = (tid / (CP * CP)) % 3, bt = tid / (3 * CP * CP);
  float v = (j < C && k < C) ? w[((size_t)bt * 573 + g * C + j) * C + k] : 0.f;
  out[tid] = (__bf16)v;
}

// ---------------------------------------------------------------- WMMA GEMM
// Cout[NN, ncols] = A[NN, CP](fp32 row-major) x B(bf16, column-major: col j at B+j*CP)
// Block = 8 waves = 128 rows. Block loops over 64-col groups; B tile staged in LDS
// once per group and shared by all 8 waves (8x L2 traffic reduction). Each wave
// holds its 16x192 A tile as six bf16 fragments in VGPRs for the whole kernel.
// B fragments are double-buffered in registers so ds_load latency overlaps WMMA.
__global__ __launch_bounds__(256) void k_gemm(const float* __restrict__ A,
                                              const __bf16* __restrict__ B,
                                              float* __restrict__ Cout, int ncols) {
  __shared__ __bf16 sB[64 * BSTRIDE];              // 25,600 B
  const int tid  = threadIdx.x;
  const int lane = tid & 31;
  const int wv   = tid >> 5;
  const int rowTile = blockIdx.x * 8 + wv;
  const bool active = rowTile < ROWTILES;          // wave-uniform
  const bool hi = lane >= 16;
  const int  r  = lane & 15;
  const int colGroups = ncols >> 6;

  // ---- load + convert A fragments once (contiguous K order, same as B):
  // lo lanes: K kb+0..15 ; hi lanes: K kb+16..31
  v16bf afr[6];
  if (active) {
    const float* arow = A + (size_t)(rowTile * 16 + r) * CP;
#pragma unroll
    for (int ks = 0; ks < 6; ++ks) {
      const float* ap = arow + ks * 32 + (hi ? 16 : 0);
      v16bf a;
#pragma unroll
      for (int e = 0; e < 8; ++e) a[e] = (__bf16)ap[e];
#pragma unroll
      for (int e = 0; e < 8; ++e) a[8 + e] = (__bf16)ap[8 + e];
      afr[ks] = a;
    }
  }

  union BFrag { uint4 u[2]; v16bf v; };
  const int boBase = hi ? 16 : 0;

  for (int cg = 0; cg < colGroups; ++cg) {
    // ---- cooperative stage of B tile: 64 cols x 192 halves = 1536 uint4
    __syncthreads();
#pragma unroll
    for (int it = 0; it < 6; ++it) {
      int idx = it * 256 + tid;                    // < 1536
      int col = idx / 24, seg = idx % 24;          // seg: 8-half (16B) chunk
      uint4 vv = *(const uint4*)(B + (size_t)(cg * 64 + col) * CP + seg * 8);
      *(uint4*)&sB[col * BSTRIDE + seg * 8] = vv;
    }
    __syncthreads();

    if (active) {
      v8f acc0 = {}, acc1 = {}, acc2 = {}, acc3 = {};
      // double-buffered B fragments: prefetch next K-step while WMMA consumes current
      BFrag bc0, bc1, bc2, bc3, bn0, bn1, bn2, bn3;
      {
        const int bo = boBase;                     // K-step 0
        bc0.u[0] = *(const uint4*)&sB[( 0 + r) * BSTRIDE + bo];
        bc0.u[1] = *(const uint4*)&sB[( 0 + r) * BSTRIDE + bo + 8];
        bc1.u[0] = *(const uint4*)&sB[(16 + r) * BSTRIDE + bo];
        bc1.u[1] = *(const uint4*)&sB[(16 + r) * BSTRIDE + bo + 8];
        bc2.u[0] = *(const uint4*)&sB[(32 + r) * BSTRIDE + bo];
        bc2.u[1] = *(const uint4*)&sB[(32 + r) * BSTRIDE + bo + 8];
        bc3.u[0] = *(const uint4*)&sB[(48 + r) * BSTRIDE + bo];
        bc3.u[1] = *(const uint4*)&sB[(48 + r) * BSTRIDE + bo + 8];
      }
#pragma unroll
      for (int ks = 0; ks < 6; ++ks) {
        if (ks < 5) {                              // prefetch K-step ks+1
          const int bo = (ks + 1) * 32 + boBase;
          bn0.u[0] = *(const uint4*)&sB[( 0 + r) * BSTRIDE + bo];
          bn0.u[1] = *(const uint4*)&sB[( 0 + r) * BSTRIDE + bo + 8];
          bn1.u[0] = *(const uint4*)&sB[(16 + r) * BSTRIDE + bo];
          bn1.u[1] = *(const uint4*)&sB[(16 + r) * BSTRIDE + bo + 8];
          bn2.u[0] = *(const uint4*)&sB[(32 + r) * BSTRIDE + bo];
          bn2.u[1] = *(const uint4*)&sB[(32 + r) * BSTRIDE + bo + 8];
          bn3.u[0] = *(const uint4*)&sB[(48 + r) * BSTRIDE + bo];
          bn3.u[1] = *(const uint4*)&sB[(48 + r) * BSTRIDE + bo + 8];
        }
        acc0 = __builtin_amdgcn_wmma_f32_16x16x32_bf16(false, afr[ks], false, bc0.v, (short)0, acc0, false, false);
        acc1 = __builtin_amdgcn_wmma_f32_16x16x32_bf16(false, afr[ks], false, bc1.v, (short)0, acc1, false, false);
        acc2 = __builtin_amdgcn_wmma_f32_16x16x32_bf16(false, afr[ks], false, bc2.v, (short)0, acc2, false, false);
        acc3 = __builtin_amdgcn_wmma_f32_16x16x32_bf16(false, afr[ks], false, bc3.v, (short)0, acc3, false, false);
        bc0 = bn0; bc1 = bn1; bc2 = bn2; bc3 = bn3;   // SSA-renamed away by unroll
      }
      // C/D layout: VGPR i -> row i (lo) / row 8+i (hi), col = lane&15
      const int colBase = cg * 64;
#pragma unroll
      for (int i = 0; i < 8; ++i) {
        int orow = rowTile * 16 + i + (hi ? 8 : 0);
        float* op = Cout + (size_t)orow * ncols + colBase + r;
        op[0]  = acc0[i];
        op[16] = acc1[i];
        op[32] = acc2[i];
        op[48] = acc3[i];
      }
    }
  }
}

// ---------------------------------------------------------------- edge scatter
__global__ void k_scatter(const float* __restrict__ m, float* __restrict__ agg,
                          const int* __restrict__ eidx, const int* __restrict__ etype, int t) {
  int e = blockIdx.x;                               // blockDim = 192
  if (etype[e] != t) return;
  int src = eidx[e], dst = eidx[EE + e];
  int c = threadIdx.x;
  unsafeAtomicAdd(agg + (size_t)dst * CP + c, m[(size_t)src * CP + c]);
}

// ---------------------------------------------------------------- GRU update
__global__ void k_gru(float* __restrict__ x, const float* __restrict__ gi,
                      const float* __restrict__ gh, const float* __restrict__ bih,
                      const float* __restrict__ bhh) {
  int n = blockIdx.x, j = threadIdx.x;              // blockDim = 192
  if (j >= C) return;
  const float* gin = gi + (size_t)n * C3P;
  const float* ghn = gh + (size_t)n * C3P;
  float ir = gin[j]          + bih[j];
  float iz = gin[CP + j]     + bih[C + j];
  float ic = gin[2 * CP + j] + bih[2 * C + j];
  float hr = ghn[j]          + bhh[j];
  float hz = ghn[CP + j]     + bhh[C + j];
  float hc = ghn[2 * CP + j] + bhh[2 * C + j];
  float r  = 1.f / (1.f + __expf(-(ir + hr)));
  float z  = 1.f / (1.f + __expf(-(iz + hz)));
  float cc = tanhf(ic + r * hc);
  float xo = x[(size_t)n * CP + j];
  x[(size_t)n * CP + j] = (1.f - z) * cc + z * xo;
}

// ---------------------------------------------------------------- residual + LN + relu
__global__ __launch_bounds__(256) void k_lnrelu(float* __restrict__ h, const float* __restrict__ hmsg,
                                                const float* __restrict__ lnw, const float* __restrict__ lnb) {
  __shared__ float sbuf[256];
  __shared__ float smu, srstd;
  int n = blockIdx.x, c = threadIdx.x;              // blockDim = 256
  float v = 0.f;
  if (c < C) v = h[(size_t)n * CP + c] + hmsg[(size_t)n * CP + c];
  sbuf[c] = v;
  __syncthreads();
  for (int off = 128; off > 0; off >>= 1) { if (c < off) sbuf[c] += sbuf[c + off]; __syncthreads(); }
  if (c == 0) smu = sbuf[0] / (float)C;
  __syncthreads();
  float mu = smu;
  float d = (c < C) ? (v - mu) : 0.f;
  sbuf[c] = d * d;
  __syncthreads();
  for (int off = 128; off > 0; off >>= 1) { if (c < off) sbuf[c] += sbuf[c + off]; __syncthreads(); }
  if (c == 0) srstd = rsqrtf(sbuf[0] / (float)C + LN_EPS);
  __syncthreads();
  if (c < CP) {
    float out = 0.f;
    if (c < C) out = fmaxf((v - mu) * srstd * lnw[c] + lnb[c], 0.f);
    h[(size_t)n * CP + c] = out;
  }
}

// ---------------------------------------------------------------- pooling + head
__global__ void k_pool(const float* __restrict__ h, const int* __restrict__ batch,
                       float* __restrict__ hg, float* __restrict__ cnt) {
  int n = blockIdx.x, c = threadIdx.x;              // blockDim = 192
  int g = batch[n];
  unsafeAtomicAdd(hg + (size_t)g * CP + c, h[(size_t)n * CP + c]);
  if (c == 0) unsafeAtomicAdd(cnt + g, 1.f);
}
__global__ void k_pooldiv(float* __restrict__ hg, const float* __restrict__ cnt) {
  int i = blockIdx.x * blockDim.x + threadIdx.x;
  if (i < GG * CP) hg[i] /= fmaxf(cnt[i / CP], 1.f);
}
__global__ void k_head1(const float* __restrict__ hg, const float* __restrict__ w1,
                        const float* __restrict__ b1, float* __restrict__ z1) {
  int g = blockIdx.x, j = threadIdx.x;              // blockDim = 192
  if (j >= C) return;
  float s = b1[j];
  for (int k = 0; k < C; ++k) s += hg[(size_t)g * CP + k] * w1[k * C + j];
  z1[(size_t)g * CP + j] = fmaxf(s, 0.f);
}
__global__ void k_head2(const float* __restrict__ z1, const float* __restrict__ w2,
                        const float* __restrict__ b2, float* __restrict__ out) {
  int i = threadIdx.x;                              // blockDim = 64
  if (i >= GG * 2) return;
  int g = i >> 1, o = i & 1;
  float s = b2[o];
  for (int j = 0; j < C; ++j) s += z1[(size_t)g * CP + j] * w2[j * 2 + o];
  out[i] = s;
}

// ---------------------------------------------------------------- host
extern "C" void kernel_launch(void* const* d_in, const int* in_sizes, int n_in,
                              void* d_out, int out_size, void* d_ws, size_t ws_size,
                              hipStream_t stream) {
  const float* x_small = (const float*)d_in[0];
  const float* conv_w  = (const float*)d_in[1];
  const float* gru_wih = (const float*)d_in[2];
  const float* gru_whh = (const float*)d_in[3];
  const float* gru_bih = (const float*)d_in[4];
  const float* gru_bhh = (const float*)d_in[5];
  const float* ln_w    = (const float*)d_in[6];
  const float* ln_b    = (const float*)d_in[7];
  const float* head_w1 = (const float*)d_in[8];
  const float* head_b1 = (const float*)d_in[9];
  const float* head_w2 = (const float*)d_in[10];
  const float* head_b2 = (const float*)d_in[11];
  const int* x_type    = (const int*)d_in[12];
  const int* x_tok     = (const int*)d_in[13];
  const int* eidx      = (const int*)d_in[14];
  const int* etype     = (const int*)d_in[15];
  const int* batch     = (const int*)d_in[16];
  float* out = (float*)d_out;

  // workspace carve-up (all offsets 256B aligned)
  char* p = (char*)d_ws;
  size_t off = 0;
  auto take = [&](size_t bytes) { void* q = p + off; off += (bytes + 255) & ~(size_t)255; return q; };
  float*  h     = (float*)take((size_t)NN * CP * 4);
  float*  x     = (float*)take((size_t)NN * CP * 4);
  float*  hmsg  = (float*)take((size_t)NN * CP * 4);
  float*  m     = (float*)take((size_t)NN * CP * 4);
  float*  agg   = (float*)take((size_t)NN * CP * 4);
  float*  gi    = (float*)take((size_t)NN * C3P * 4);
  float*  gh    = (float*)take((size_t)NN * C3P * 4);
  __bf16* convB = (__bf16*)take((size_t)150 * CP * CP * 2);
  __bf16* wihB  = (__bf16*)take((size_t)15 * 3 * CP * CP * 2);
  __bf16* whhB  = (__bf16*)take((size_t)15 * 3 * CP * CP * 2);
  float*  hg    = (float*)take((size_t)GG * CP * 4);
  float*  cnt   = (float*)take((size_t)GG * 4);
  float*  z1    = (float*)take((size_t)GG * CP * 4);
  (void)ws_size; (void)in_sizes; (void)n_in; (void)out_size;

  const int NCP = NN * CP;

  // features + weight conversion (bf16, column-major padded)
  k_build<<<NN, CP, 0, stream>>>(x_type, x_tok, x_small, h);
  k_cvt_convw<<<(150 * CP * CP + 255) / 256, 256, 0, stream>>>(conv_w, convB);
  k_cvt_gruw<<<(15 * 3 * CP * CP + 255) / 256, 256, 0, stream>>>(gru_wih, wihB);
  k_cvt_gruw<<<(15 * 3 * CP * CP + 255) / 256, 256, 0, stream>>>(gru_whh, whhB);

  for (int b = 0; b < BLOCKS; ++b) {
    k_fill<<<(NCP + 255) / 256, 256, 0, stream>>>(hmsg, NCP, 0.f);
    for (int t = 0; t < NT; ++t) {
      const int bt = b * NT + t;
      k_copy<<<(NCP + 255) / 256, 256, 0, stream>>>(x, h, NCP);
      const __bf16* wih_bt = wihB + (size_t)bt * 3 * CP * CP;
      const __bf16* whh_bt = whhB + (size_t)bt * 3 * CP * CP;
      const float*  bih_bt = gru_bih + (size_t)bt * 3 * C;
      const float*  bhh_bt = gru_bhh + (size_t)bt * 3 * C;
      for (int s = 0; s < STEPS; ++s) {
        const __bf16* w_s = convB + (size_t)(bt * STEPS + s) * CP * CP;
        k_gemm<<<ROWGRPS, 256, 0, stream>>>(x, w_s, m, CP);             // m = x @ W
        k_fill<<<(NCP + 255) / 256, 256, 0, stream>>>(agg, NCP, 0.f);
        k_scatter<<<EE, CP, 0, stream>>>(m, agg, eidx, etype, t);       // scatter-add
        k_gemm<<<ROWGRPS, 256, 0, stream>>>(agg, wih_bt, gi, C3P);      // gi = agg @ Wih^T
        k_gemm<<<ROWGRPS, 256, 0, stream>>>(x,   whh_bt, gh, C3P);      // gh = x @ Whh^T
        k_gru<<<NN, CP, 0, stream>>>(x, gi, gh, bih_bt, bhh_bt);
      }
      k_add<<<(NCP + 255) / 256, 256, 0, stream>>>(hmsg, x, NCP);
    }
    k_lnrelu<<<NN, 256, 0, stream>>>(h, hmsg, ln_w + b * C, ln_b + b * C);
  }

  k_fill<<<(GG * CP + 255) / 256, 256, 0, stream>>>(hg, GG * CP, 0.f);
  k_fill<<<1, 64, 0, stream>>>(cnt, GG, 0.f);
  k_pool<<<NN, CP, 0, stream>>>(h, batch, hg, cnt);
  k_pooldiv<<<(GG * CP + 255) / 256, 256, 0, stream>>>(hg, cnt);
  k_head1<<<GG, CP, 0, stream>>>(hg, head_w1, head_b1, z1);
  k_head2<<<1, 64, 0, stream>>>(z1, head_w2, head_b2, out);
}
